// NormLinearAttention_65231963291665
// MI455X (gfx1250) — compile-verified, matrix operand order checked by reading, not run on hardware
//
#include <hip/hip_runtime.h>
#include <hip/hip_bf16.h>

typedef __bf16 bf16_t;
typedef __attribute__((ext_vector_type(16))) __bf16 v16bf;
typedef __attribute__((ext_vector_type(8)))  float  v8f;

#define WMMA_BF16(A_, B_, C_) \
  __builtin_amdgcn_wmma_f32_16x16x32_bf16(false, (A_), false, (B_), (short)0, (C_), false, false)

// --- CDNA5 async global->LDS path (ASYNCcnt), guarded so compile never breaks
#if defined(__has_builtin)
#if __has_builtin(__builtin_amdgcn_global_load_async_to_lds_b128) && \
    __has_builtin(__builtin_amdgcn_s_wait_asynccnt)
#define HAVE_ASYNC_LDS 1
#endif
#endif
#ifndef HAVE_ASYNC_LDS
#define HAVE_ASYNC_LDS 0
#endif

#if HAVE_ASYNC_LDS
typedef int v4i_raw __attribute__((vector_size(16)));
typedef __attribute__((address_space(1))) v4i_raw gas_v4i;  // global <4 x i32>
typedef __attribute__((address_space(3))) v4i_raw las_v4i;  // LDS    <4 x i32>
__device__ __forceinline__ void async_b128(const void* g, void* l) {
  // low 32 bits of a generic LDS pointer are the LDS byte offset
  __builtin_amdgcn_global_load_async_to_lds_b128(
      (gas_v4i*)(uintptr_t)g, (las_v4i*)(uint32_t)(uintptr_t)l, 0, 0);
}
#endif

// ---------------------------------------------------------------------------
// Kernel 1: f32 -> bf16 cast
// ---------------------------------------------------------------------------
__global__ __launch_bounds__(256) void cast_f32_bf16(const float* __restrict__ src,
                                                     bf16_t* __restrict__ dst, int n) {
  int i = blockIdx.x * blockDim.x + threadIdx.x;
  int stride = gridDim.x * blockDim.x;
  for (; i < n; i += stride) dst[i] = (bf16_t)src[i];
}

// ---------------------------------------------------------------------------
// Kernel 2/5: bf16 WMMA GEMM  C[M,N] = A[M,K] * B[N,K]^T
// double-buffered LDS, async global->LDS staging when available
// mode 0: qkvu epilogue (silu on q/k, scatter to head layout, u -> f32)
// mode 1: plain f32 store to outF [M,N]
// ---------------------------------------------------------------------------
__device__ __forceinline__ float silu_f(float x) {
  return x / (1.0f + __expf(-x));
}

// stage one 128x32 A tile + 128x32 B tile into LDS (4 b128 ops per thread)
__device__ __forceinline__ void stage_tile(const bf16_t* __restrict__ A,
                                           const bf16_t* __restrict__ B,
                                           int m0, int n0, int K, int k0,
                                           bf16_t (*as)[40], bf16_t (*bs)[40],
                                           int r, int c) {
  for (int rr = 0; rr < 2; ++rr) {
    int row = r + rr * 64;
    const bf16_t* ga = A + (size_t)(m0 + row) * K + k0 + c * 8;
    const bf16_t* gb = B + (size_t)(n0 + row) * K + k0 + c * 8;
#if HAVE_ASYNC_LDS
    async_b128(ga, &as[row][c * 8]);
    async_b128(gb, &bs[row][c * 8]);
#else
    *(uint4*)&as[row][c * 8] = *(const uint4*)ga;
    *(uint4*)&bs[row][c * 8] = *(const uint4*)gb;
#endif
  }
}

__global__ __launch_bounds__(256) void gemm_bf16_wmma(
    const bf16_t* __restrict__ A, const bf16_t* __restrict__ B,
    int M, int N, int K, int mode,
    float* __restrict__ outF,
    bf16_t* __restrict__ qbuf, bf16_t* __restrict__ kbuf,
    bf16_t* __restrict__ vbuf, float* __restrict__ ubuf) {
  __shared__ bf16_t As[2][128][40];  // padded stride (80B) to spread LDS banks
  __shared__ bf16_t Bs[2][128][40];

  const int t = threadIdx.x;
  const int n0 = blockIdx.x * 128;
  const int m0 = blockIdx.y * 128;

  const int w = t >> 5;         // wave 0..7
  const int lane = t & 31;
  const int lr = lane & 15;     // fragment row/col within tile
  const int lh = lane >> 4;     // K-half selector
  const int waveM = w >> 2;     // 0..1  -> 64 rows
  const int waveN = w & 3;      // 0..3  -> 32 cols

  v8f acc[4][2];
  for (int i = 0; i < 4; ++i)
    for (int j = 0; j < 2; ++j)
      acc[i][j] = (v8f){0.f, 0.f, 0.f, 0.f, 0.f, 0.f, 0.f, 0.f};

  const int r = t >> 2;  // 0..63
  const int c = t & 3;   // 0..3 (chunks of 8 bf16)
  const int NK = K >> 5;

  // prologue: tile 0 into buffer 0
  stage_tile(A, B, m0, n0, K, 0, As[0], Bs[0], r, c);

  for (int kt = 0; kt < NK; ++kt) {
    const int cur = kt & 1;
    if (kt + 1 < NK) {
      stage_tile(A, B, m0, n0, K, (kt + 1) << 5, As[cur ^ 1], Bs[cur ^ 1], r, c);
#if HAVE_ASYNC_LDS
      __builtin_amdgcn_s_wait_asynccnt(4);  // tile kt done; tile kt+1 in flight
#endif
    } else {
#if HAVE_ASYNC_LDS
      __builtin_amdgcn_s_wait_asynccnt(0);
#endif
    }
    __syncthreads();

    v16bf af[4], bfr[2];
    for (int i = 0; i < 4; ++i)
      af[i] = *(const v16bf*)&As[cur][waveM * 64 + i * 16 + lr][lh * 16];
    for (int j = 0; j < 2; ++j)
      bfr[j] = *(const v16bf*)&Bs[cur][waveN * 32 + j * 16 + lr][lh * 16];

    for (int i = 0; i < 4; ++i)
      for (int j = 0; j < 2; ++j)
        acc[i][j] = WMMA_BF16(af[i], bfr[j], acc[i][j]);
    __syncthreads();  // release buf[cur] before it is re-staged
  }

  // epilogue
  for (int i = 0; i < 4; ++i) {
    for (int j = 0; j < 2; ++j) {
      v8f cacc = acc[i][j];
      for (int e = 0; e < 8; ++e) {
        int m = m0 + waveM * 64 + i * 16 + e + lh * 8;
        int n = n0 + waveN * 32 + j * 16 + lr;
        float val = cacc[e];
        if (mode == 1) {
          outF[(size_t)m * N + n] = val;
        } else {
          int seg = n >> 11;          // 0:q 1:k 2:v 3:u
          int hh = (n >> 7) & 15;     // head
          int dd = n & 127;           // dim within head
          int bidx = m >> 12;         // batch
          int tok = m & 4095;         // token
          size_t hidx = (((size_t)(bidx * 16 + hh)) * 4096 + tok) * 128 + dd;
          if (seg == 0)      qbuf[hidx] = (bf16_t)silu_f(val);
          else if (seg == 1) kbuf[hidx] = (bf16_t)silu_f(val);
          else if (seg == 2) vbuf[hidx] = (bf16_t)val;
          else ubuf[((size_t)bidx * 4096 + tok) * 2048 + (n & 2047)] = val;
        }
      }
    }
  }
}

// ---------------------------------------------------------------------------
// Kernel 3: block-wise linear attention scan, one workgroup per (b,h)
// LDS layout (dynamic):
//   kvF  : f32  [e=128][d=128]                  65536 B
//   kvB  : bf16 [e=128][d stride 136]           34816 B
//   kiT  : bf16 [d=128][s stride 264] (k^T * k_decay)  67584 B
//   viT  : bf16 [e=128][s stride 264] (v^T)     67584 B
//   qkS  : bf16 per-wave strips 8 * [16][264]   67584 B
// total 303104 B
// ---------------------------------------------------------------------------
#define LATTN_SMEM 303104

__global__ __launch_bounds__(256) void linattn_wmma(
    const bf16_t* __restrict__ q, const bf16_t* __restrict__ k,
    const bf16_t* __restrict__ v, const float* __restrict__ slope,
    float* __restrict__ o) {
  extern __shared__ char smem[];
  float*  kvF = (float*)smem;                        // [128*128]
  bf16_t* kvB = (bf16_t*)(smem + 65536);             // [128*136]
  bf16_t* kiT = (bf16_t*)(smem + 65536 + 34816);     // [128*264]
  bf16_t* viT = kiT + 128 * 264;
  bf16_t* qkS = viT + 128 * 264;                     // 8 * 16 * 264

  const int t = threadIdx.x;
  const int bh = blockIdx.x;                 // 0..31  -> (b*16 + h)
  const int bidx = bh >> 4;
  const int hidx = bh & 15;
  const float rate = slope[hidx];
  const float bd = __expf(-rate * 256.0f);   // block decay
  const size_t base = (size_t)bh * 4096 * 128;

  const int w = t >> 5, lane = t & 31;
  const int lr = lane & 15, lh = lane >> 4;
  bf16_t* myqk = qkS + w * 16 * 264;

  // zero kv state
  for (int i = t; i < 128 * 128; i += 256) kvF[i] = 0.0f;
  for (int i = t; i < 128 * 136; i += 256) kvB[i] = (bf16_t)0.0f;
  __syncthreads();

  for (int bi = 0; bi < 16; ++bi) {
    // ---- stage 1: build kiT (scaled by k_decay) and viT, transposed ----
    {
      int s = t;  // 256 threads, one source row each
      const bf16_t* krow = k + base + (size_t)(bi * 256 + s) * 128;
      const bf16_t* vrow = v + base + (size_t)(bi * 256 + s) * 128;
      float kdec = __expf(-rate * (float)(255 - s));
      for (int dc = 0; dc < 16; ++dc) {
        uint4 kd4 = *(const uint4*)(krow + dc * 8);
        uint4 vd4 = *(const uint4*)(vrow + dc * 8);
        const bf16_t* kp = (const bf16_t*)&kd4;
        const bf16_t* vp = (const bf16_t*)&vd4;
        for (int e = 0; e < 8; ++e) {
          int d = dc * 8 + e;
          kiT[d * 264 + s] = (bf16_t)((float)kp[e] * kdec);
          viT[d * 264 + s] = vp[e];
        }
      }
    }
    __syncthreads();

    // ---- stages 2-4: per-wave output strips ----
    for (int sp = 0; sp < 2; ++sp) {
      const int strip = w * 2 + sp;  // 0..15 (16 rows each)
      // raw q fragments for this strip (A operands, K = d = 128)
      v16bf qf[4];
      for (int kk = 0; kk < 4; ++kk)
        qf[kk] = *(const v16bf*)(q + base +
                                 (size_t)(bi * 256 + strip * 16 + lr) * 128 +
                                 kk * 32 + lh * 16);

      // stage 3: qk = qi @ ki^T, masked by causal decay, cast to bf16 in LDS
      for (int st = 0; st < 16; ++st) {
        v8f a = (v8f){0.f, 0.f, 0.f, 0.f, 0.f, 0.f, 0.f, 0.f};
        for (int kk = 0; kk < 4; ++kk) {
          v16bf bk = *(const v16bf*)(k + base +
                                     (size_t)(bi * 256 + st * 16 + lr) * 128 +
                                     kk * 32 + lh * 16);
          a = WMMA_BF16(qf[kk], bk, a);
        }
        for (int e = 0; e < 8; ++e) {
          int m = strip * 16 + e + lh * 8;
          int j = st * 16 + lr;
          float dv = (m >= j) ? __expf(-rate * (float)(m - j)) : 0.0f;
          myqk[(e + lh * 8) * 264 + j] = (bf16_t)(a[e] * dv);
        }
      }

      // scale q fragments by q_decay (per-row scalar)
      {
        float qdec = __expf(-rate * (float)(strip * 16 + lr + 1));
        for (int kk = 0; kk < 4; ++kk)
          for (int e = 0; e < 16; ++e)
            qf[kk][e] = (bf16_t)((float)qf[kk][e] * qdec);
      }

      // stage 2 + 4: o = (q*qdec) @ kv  +  qk_masked @ v
      for (int et = 0; et < 8; ++et) {
        v8f oacc = (v8f){0.f, 0.f, 0.f, 0.f, 0.f, 0.f, 0.f, 0.f};
        for (int kk = 0; kk < 4; ++kk) {
          v16bf bkv = *(const v16bf*)&kvB[(et * 16 + lr) * 136 + kk * 32 + lh * 16];
          oacc = WMMA_BF16(qf[kk], bkv, oacc);
        }
        for (int sc = 0; sc < 8; ++sc) {
          v16bf aq = *(const v16bf*)&myqk[lr * 264 + sc * 32 + lh * 16];
          v16bf bv = *(const v16bf*)&viT[(et * 16 + lr) * 264 + sc * 32 + lh * 16];
          oacc = WMMA_BF16(aq, bv, oacc);
        }
        // store o rows: layout [b][n][h*128]
        for (int e = 0; e < 8; ++e) {
          int m = strip * 16 + e + lh * 8;
          int tok = bi * 256 + m;
          o[((size_t)bidx * 4096 + tok) * 2048 + hidx * 128 + et * 16 + lr] = oacc[e];
        }
      }
    }
    __syncthreads();

    // ---- stage 5: kv <- bd*kv + (ki*k_decay)^T @ vi ----
    {
      const int dt = w;  // wave w owns d-tile w
      for (int et = 0; et < 8; ++et) {
        v8f dl = (v8f){0.f, 0.f, 0.f, 0.f, 0.f, 0.f, 0.f, 0.f};
        for (int sc = 0; sc < 8; ++sc) {
          v16bf ak = *(const v16bf*)&kiT[(dt * 16 + lr) * 264 + sc * 32 + lh * 16];
          v16bf bv = *(const v16bf*)&viT[(et * 16 + lr) * 264 + sc * 32 + lh * 16];
          dl = WMMA_BF16(ak, bv, dl);
        }
        for (int e = 0; e < 8; ++e) {
          int drow = dt * 16 + e + lh * 8;
          int ecol = et * 16 + lr;
          float nv = bd * kvF[ecol * 128 + drow] + dl[e];
          kvF[ecol * 128 + drow] = nv;
          kvB[ecol * 136 + drow] = (bf16_t)nv;
        }
      }
    }
    __syncthreads();
  }
}

// ---------------------------------------------------------------------------
// Kernel 4: RMSNorm (no scale) + u gating, cast to bf16
// one block per token row, 2048 cols
// ---------------------------------------------------------------------------
__global__ __launch_bounds__(256) void rmsnorm_gate(
    const float* __restrict__ o, const float* __restrict__ u,
    bf16_t* __restrict__ out) {
  __shared__ float red[256];
  const int row = blockIdx.x;
  const int t = threadIdx.x;
  const size_t rb = (size_t)row * 2048;

  float ss = 0.0f;
  for (int i = t; i < 2048; i += 256) {
    float x = o[rb + i];
    ss += x * x;
  }
  red[t] = ss;
  __syncthreads();
  for (int s = 128; s > 0; s >>= 1) {
    if (t < s) red[t] += red[t + s];
    __syncthreads();
  }
  const float scale = rsqrtf(red[0] * (1.0f / 2048.0f) + 1e-6f);
  for (int i = t; i < 2048; i += 256) {
    out[rb + i] = (bf16_t)(o[rb + i] * scale * u[rb + i]);
  }
}

// ---------------------------------------------------------------------------
// host launcher
// ---------------------------------------------------------------------------
extern "C" void kernel_launch(void* const* d_in, const int* in_sizes, int n_in,
                              void* d_out, int out_size, void* d_ws, size_t ws_size,
                              hipStream_t stream) {
  (void)in_sizes; (void)n_in; (void)out_size; (void)ws_size;
  const float* x      = (const float*)d_in[0];
  const float* slope  = (const float*)d_in[1];
  const float* Wqkvu  = (const float*)d_in[2];
  const float* Wout   = (const float*)d_in[3];
  float* outF = (float*)d_out;

  const int M = 2 * 4096;     // 8192 tokens
  const int E = 2048;         // embed / hidden
  const int N4 = 4 * E;       // 8192

  size_t off = 0;
  auto wsp = [&](size_t bytes) -> void* {
    void* p = (char*)d_ws + off;
    off += (bytes + 255) & ~(size_t)255;
    return p;
  };
  bf16_t* xb    = (bf16_t*)wsp((size_t)M * E * 2);       // 32 MB
  bf16_t* wqb   = (bf16_t*)wsp((size_t)N4 * E * 2);      // 32 MB
  bf16_t* wob   = (bf16_t*)wsp((size_t)E * E * 2);       //  8 MB
  bf16_t* qb    = (bf16_t*)wsp((size_t)M * E * 2);       // 32 MB [b,h,n,d]
  bf16_t* kb    = (bf16_t*)wsp((size_t)M * E * 2);       // 32 MB
  bf16_t* vb    = (bf16_t*)wsp((size_t)M * E * 2);       // 32 MB
  float*  ub    = (float*)wsp((size_t)M * E * 4);        // 64 MB [b,n,hid]
  float*  ob    = (float*)wsp((size_t)M * E * 4);        // 64 MB [b,n,hid]
  bf16_t* onb   = (bf16_t*)wsp((size_t)M * E * 2);       // 32 MB

  // 1) casts
  cast_f32_bf16<<<2048, 256, 0, stream>>>(x, xb, M * E);
  cast_f32_bf16<<<2048, 256, 0, stream>>>(Wqkvu, wqb, N4 * E);
  cast_f32_bf16<<<1024, 256, 0, stream>>>(Wout, wob, E * E);

  // 2) qkvu GEMM with fused silu/split epilogue
  gemm_bf16_wmma<<<dim3(N4 / 128, M / 128), 256, 0, stream>>>(
      xb, wqb, M, N4, E, /*mode=*/0, nullptr, qb, kb, vb, ub);

  // 3) linear attention scan (one WG per b*h)
  (void)hipFuncSetAttribute((const void*)linattn_wmma,
                            hipFuncAttributeMaxDynamicSharedMemorySize,
                            LATTN_SMEM);
  linattn_wmma<<<32, 256, LATTN_SMEM, stream>>>(qb, kb, vb, slope, ob);

  // 4) RMSNorm + u gating
  rmsnorm_gate<<<M, 256, 0, stream>>>(ob, ub, onb);

  // 5) output GEMM
  gemm_bf16_wmma<<<dim3(E / 128, M / 128), 256, 0, stream>>>(
      onb, wob, M, E, E, /*mode=*/1, outF, nullptr, nullptr, nullptr, nullptr);
}
